// EdgeNet_83708912599063
// MI455X (gfx1250) — matrix operand order repaired and checked by reading, not compiled, not run on hardware
//
#include <hip/hip_runtime.h>
#include <hip/hip_bf16.h>
#include <cstdint>

// ---------------------------------------------------------------------------
// EdgeConv GNN on MI455X (gfx1250, wave32).
// Edge MLPs run as f16 WMMA GEMMs (v_wmma_f32_16x16x32_f16), 16 edges/tile/wave.
// Node-side work is scalar (tiny). Segment-sum via global_atomic_add_f32.
// ---------------------------------------------------------------------------

typedef __attribute__((ext_vector_type(16))) _Float16 v16h;
typedef __attribute__((ext_vector_type(8)))  _Float16 v8h;
typedef __attribute__((ext_vector_type(8)))  float    v8f;

#define HIDN   32
#define FPITCH 40      // halfs per node-feature row (35 used, padded to 80B)
#define KPAD   96      // K padded to 3 WMMA slices of 32
#define NWAVES 8       // waves per 256-thread block

// ---- WMMA fragment loaders (layouts per CDNA5 ISA 7.12.2, wave32) ----------
// A (16x32 f16): lane holds row m=lane%16; elements = halfs [kb+hs*8,+8) and
// [kb+16+hs*8,+8)  (hs = lane/16).
__device__ __forceinline__ v16h lds_frag_a(const _Float16* rowp, int kbase, int hs) {
  union { v16h v; v8h h[2]; } u;
  u.h[0] = *(const v8h*)(rowp + kbase + hs * 8);
  u.h[1] = *(const v8h*)(rowp + kbase + 16 + hs * 8);
  return u.v;
}
// B (32x16 f16): lane holds column n=lane%16; elements = halfs [kb+hs*8,+8) and
// [kb+16+hs*8,+8) (weight tile stored N-major: row n = 96/64 contiguous K).
__device__ __forceinline__ v16h lds_frag_b(const _Float16* rowp, int kbase, int hs) {
  union { v16h v; v8h h[2]; } u;
  const _Float16* p = rowp + kbase + hs * 8;
  u.h[0] = *(const v8h*)(p);
  u.h[1] = *(const v8h*)(p + 16);
  return u.v;
}

__device__ __forceinline__ v8f wmma16x16x32(v16h a, v16h b, v8f c) {
  return __builtin_amdgcn_wmma_f32_16x16x32_f16(false, a, false, b, (short)0, c,
                                                false, false);
}

// ---------------------------------------------------------------------------
// Workspace zero (BN stats accumulators + segment-sum buffer)
// ---------------------------------------------------------------------------
__global__ void __launch_bounds__(256) k_zero(float* stats, float* Hsum, long long total) {
  long long i = (long long)blockIdx.x * blockDim.x + threadIdx.x;
  if (i < 64) stats[i] = 0.f;
  if (i < total) Hsum[i] = 0.f;
}

// ---------------------------------------------------------------------------
// BatchNorm batch statistics: per-feature sum / sumsq -> stats[0..5]
// ---------------------------------------------------------------------------
__global__ void __launch_bounds__(256) k_stats(const float* __restrict__ x,
                                               float* stats, int N) {
  float s[3] = {0.f, 0.f, 0.f}, q[3] = {0.f, 0.f, 0.f};
  for (int n = blockIdx.x * blockDim.x + threadIdx.x; n < N;
       n += gridDim.x * blockDim.x) {
#pragma unroll
    for (int d = 0; d < 3; d++) { float v = x[n * 3 + d]; s[d] += v; q[d] += v * v; }
  }
#pragma unroll
  for (int d = 0; d < 3; d++) {
#pragma unroll
    for (int off = 16; off > 0; off >>= 1) {
      s[d] += __shfl_down(s[d], off, 32);
      q[d] += __shfl_down(q[d], off, 32);
    }
  }
  if ((threadIdx.x & 31) == 0) {
#pragma unroll
    for (int d = 0; d < 3; d++) {
      atomicAdd(&stats[d], s[d]);
      atomicAdd(&stats[3 + d], q[d]);
    }
  }
}

__global__ void k_finalize(float* stats, int N) {
  int d = threadIdx.x;
  if (d < 3) {
    float mu  = stats[d] / (float)N;
    float var = stats[3 + d] / (float)N - mu * mu;
    stats[6 + d] = mu;
    stats[9 + d] = rsqrtf(var + 1e-5f);
  }
}

// ---------------------------------------------------------------------------
// Node features: X = BN(x); H = tanh(relu(X@w1+b1)@w2+b2); F1 = [H(32), X(3)]
// packed as f16 rows of FPITCH halfs.
// ---------------------------------------------------------------------------
__global__ void __launch_bounds__(256) k_nodes(
    const float* __restrict__ x, const float* __restrict__ stats,
    const float* __restrict__ bn_g, const float* __restrict__ bn_b,
    const float* __restrict__ w1, const float* __restrict__ b1,
    const float* __restrict__ w2, const float* __restrict__ b2,
    _Float16* __restrict__ F1, int N) {
  __shared__ float sW1[96], sB1[32], sW2[1024], sB2[32], sMU[3], sRS[3], sG[3], sBB[3];
  int tid = threadIdx.x;
  for (int i = tid; i < 96; i += 256) sW1[i] = w1[i];
  for (int i = tid; i < 1024; i += 256) sW2[i] = w2[i];
  if (tid < 32) { sB1[tid] = b1[tid]; sB2[tid] = b2[tid]; }
  if (tid < 3) { sMU[tid] = stats[6 + tid]; sRS[tid] = stats[9 + tid];
                 sG[tid] = bn_g[tid];       sBB[tid] = bn_b[tid]; }
  __syncthreads();
  int n = blockIdx.x * 256 + tid;
  if (n >= N) return;
  float X[3];
#pragma unroll
  for (int d = 0; d < 3; d++)
    X[d] = (x[n * 3 + d] - sMU[d]) * sRS[d] * sG[d] + sBB[d];
  float h[32];
#pragma unroll
  for (int j = 0; j < 32; j++) {
    float s = sB1[j] + X[0] * sW1[j] + X[1] * sW1[32 + j] + X[2] * sW1[64 + j];
    h[j] = s > 0.f ? s : 0.f;
  }
  _Float16* out = F1 + (long long)n * FPITCH;
  for (int j2 = 0; j2 < 32; j2++) {
    float s = sB2[j2];
#pragma unroll
    for (int j = 0; j < 32; j++) s += h[j] * sW2[j * 32 + j2];
    out[j2] = (_Float16)tanhf(s);
  }
  out[32] = (_Float16)X[0]; out[33] = (_Float16)X[1]; out[34] = (_Float16)X[2];
  out[35] = (_Float16)0.f; out[36] = (_Float16)0.f; out[37] = (_Float16)0.f;
  out[38] = (_Float16)0.f; out[39] = (_Float16)0.f;
}

// ---------------------------------------------------------------------------
// EdgeConv message GEMM + segment-sum.
// A row (K=96) = [x_i(0..34), 0, x_j(36..70), ea(71), 0...]; the (x_j - x_i)
// term is folded into W': W'[k]=W[k]-W[35+k] (k<35), W'[35]=0, W'[36+k]=W[35+k].
// ---------------------------------------------------------------------------
__global__ void __launch_bounds__(256) k_edgeconv(
    const _Float16* __restrict__ F1, const long long* __restrict__ ei,
    const float* __restrict__ eattr,
    const float* __restrict__ w_c1, const float* __restrict__ b_c1,
    const float* __restrict__ w_c2, const float* __restrict__ b_c2,
    float* __restrict__ Hsum, long long E, int ntiles) {
  __shared__ __align__(16) _Float16 W1t[64][KPAD];          // N-major, K padded
  __shared__ __align__(16) _Float16 W2t[32][64];
  __shared__ __align__(16) _Float16 Atile[NWAVES][16][KPAD];
  __shared__ __align__(16) _Float16 Htile[NWAVES][16][64];
  __shared__ float B1s[64];
  __shared__ float B2s[32];
  __shared__ int   colbuf[NWAVES][16];

  const int tid = threadIdx.x;
  for (int i = tid; i < 64 * KPAD; i += 256) {
    int n = i / KPAD, k = i % KPAD;
    float v = 0.f;
    if (k < 35)                 v = w_c1[k * 64 + n] - w_c1[(35 + k) * 64 + n];
    else if (k >= 36 && k < 71) v = w_c1[(k - 1) * 64 + n];
    else if (k == 71)           v = w_c1[70 * 64 + n];
    W1t[n][k] = (_Float16)v;
  }
  for (int i = tid; i < 32 * 64; i += 256) {
    int n = i / 64, k = i % 64;
    W2t[n][k] = (_Float16)w_c2[k * 32 + n];
  }
  if (tid < 64) B1s[tid] = b_c1[tid];
  if (tid < 32) B2s[tid] = b_c2[tid];
  for (int i = tid; i < NWAVES * 16 * KPAD; i += 256)
    ((_Float16*)Atile)[i] = (_Float16)0.f;   // pad columns must stay exactly 0
  __syncthreads();

  const int w = tid >> 5, lane = tid & 31, ln = lane & 15, hs = lane >> 4;
  const long long* rowIdx = ei;        // edge_index[0] = source (x_j)
  const long long* colIdx = ei + E;    // edge_index[1] = target (x_i), aggregate

  // Weight B-fragments live in registers for the whole kernel.
  v16h bC1[4][3];
#pragma unroll
  for (int nt = 0; nt < 4; nt++)
#pragma unroll
    for (int s = 0; s < 3; s++)
      bC1[nt][s] = lds_frag_b(&W1t[nt * 16 + ln][0], s * 32, hs);
  v16h bC2[2][2];
#pragma unroll
  for (int nt = 0; nt < 2; nt++)
#pragma unroll
    for (int s = 0; s < 2; s++)
      bC2[nt][s] = lds_frag_b(&W2t[nt * 16 + ln][0], s * 32, hs);

  for (int t = blockIdx.x * NWAVES + w; t < ntiles; t += gridDim.x * NWAVES) {
    const long long tb = (long long)t * 16;
    {
      long long e = tb + ln; if (e >= E) e = E - 1;   // tail: duplicate, masked later
      if (hs == 0) {                                   // x_i = F1[col]
        long long ni = colIdx[e];
        colbuf[w][ln] = (int)ni;
        const unsigned int* src = (const unsigned int*)(F1 + ni * FPITCH);
        unsigned int* dst = (unsigned int*)&Atile[w][ln][0];
#pragma unroll
        for (int i = 0; i < 17; i++) dst[i] = src[i];
        Atile[w][ln][34] = F1[ni * FPITCH + 34];
      } else {                                         // x_j = F1[row]
        long long nj = rowIdx[e];
        const unsigned int* src = (const unsigned int*)(F1 + nj * FPITCH);
        unsigned int* dst = (unsigned int*)&Atile[w][ln][36];
#pragma unroll
        for (int i = 0; i < 17; i++) dst[i] = src[i];
        Atile[w][ln][70] = F1[nj * FPITCH + 34];
        Atile[w][ln][71] = (_Float16)eattr[e];
      }
    }
    asm volatile("s_wait_dscnt 0" ::: "memory");  // DS in-order per wave: stores visible

    const _Float16* arow = &Atile[w][ln][0];
    v16h a0 = lds_frag_a(arow, 0, hs);
    v16h a1 = lds_frag_a(arow, 32, hs);
    v16h a2 = lds_frag_a(arow, 64, hs);

#pragma unroll
    for (int nt = 0; nt < 4; nt++) {                   // hidden = relu(A @ W1' + b1)
      v8f c = {0.f, 0.f, 0.f, 0.f, 0.f, 0.f, 0.f, 0.f};
      c = wmma16x16x32(a0, bC1[nt][0], c);
      c = wmma16x16x32(a1, bC1[nt][1], c);
      c = wmma16x16x32(a2, bC1[nt][2], c);
      const int cix = nt * 16 + ln;
      const float bb = B1s[cix];
#pragma unroll
      for (int r = 0; r < 8; r++) {                    // C layout: M = r + 8*hs, N = cix
        float v = c[r] + bb;
        Htile[w][r + 8 * hs][cix] = (_Float16)(v > 0.f ? v : 0.f);
      }
    }
    asm volatile("s_wait_dscnt 0" ::: "memory");

    const _Float16* hrow = &Htile[w][ln][0];
    v16h h0 = lds_frag_a(hrow, 0, hs);
    v16h h1 = lds_frag_a(hrow, 32, hs);
#pragma unroll
    for (int nt = 0; nt < 2; nt++) {                   // msg = tanh(hidden @ W2 + b2)
      v8f c = {0.f, 0.f, 0.f, 0.f, 0.f, 0.f, 0.f, 0.f};
      c = wmma16x16x32(h0, bC2[nt][0], c);
      c = wmma16x16x32(h1, bC2[nt][1], c);
      const int cix = nt * 16 + ln;
      const float bb = B2s[cix];
#pragma unroll
      for (int r = 0; r < 8; r++) {
        long long e = tb + r + 8 * hs;
        if (e < E) {
          float v = tanhf(c[r] + bb);
          atomicAdd(&Hsum[(long long)colbuf[w][r + 8 * hs] * HIDN + cix], v);
        }
      }
    }
  }
}

// ---------------------------------------------------------------------------
// F2 = [segment_sum H (32), X (3)] packed to f16
// ---------------------------------------------------------------------------
__global__ void __launch_bounds__(256) k_f2(
    const float* __restrict__ Hsum, const float* __restrict__ x,
    const float* __restrict__ stats, const float* __restrict__ bn_g,
    const float* __restrict__ bn_b, _Float16* __restrict__ F2, int N) {
  int n = blockIdx.x * 256 + threadIdx.x;
  if (n >= N) return;
  _Float16* out = F2 + (long long)n * FPITCH;
  const float* hrow = Hsum + (long long)n * HIDN;
#pragma unroll
  for (int k = 0; k < HIDN; k++) out[k] = (_Float16)hrow[k];
#pragma unroll
  for (int d = 0; d < 3; d++)
    out[32 + d] = (_Float16)((x[n * 3 + d] - stats[6 + d]) * stats[9 + d] * bn_g[d] + bn_b[d]);
}

// ---------------------------------------------------------------------------
// Output head: sigmoid(relu([xc[row], xc[col], ea] @ w_e1 + b_e1) @ w_e2 + b_e2)
// Same WMMA tiling; no subtraction fold; layer-2 (N=1) is a per-edge dot.
// ---------------------------------------------------------------------------
__global__ void __launch_bounds__(256) k_edgeout(
    const _Float16* __restrict__ F2, const long long* __restrict__ ei,
    const float* __restrict__ eattr,
    const float* __restrict__ w_e1, const float* __restrict__ b_e1,
    const float* __restrict__ w_e2, const float* __restrict__ b_e2,
    float* __restrict__ out, long long E, int ntiles) {
  __shared__ __align__(16) _Float16 W1t[64][KPAD];
  __shared__ __align__(16) _Float16 Atile[NWAVES][16][KPAD];
  __shared__ __align__(16) _Float16 Htile[NWAVES][16][64];
  __shared__ float B1s[64];
  __shared__ float We2[64];
  __shared__ float Be2;

  const int tid = threadIdx.x;
  for (int i = tid; i < 64 * KPAD; i += 256) {
    int n = i / KPAD, k = i % KPAD;
    float v = 0.f;
    if (k < 35)                 v = w_e1[k * 64 + n];
    else if (k >= 36 && k < 71) v = w_e1[(k - 1) * 64 + n];
    else if (k == 71)           v = w_e1[70 * 64 + n];
    W1t[n][k] = (_Float16)v;
  }
  if (tid < 64) { B1s[tid] = b_e1[tid]; We2[tid] = w_e2[tid]; }
  if (tid == 0) Be2 = b_e2[0];
  for (int i = tid; i < NWAVES * 16 * KPAD; i += 256)
    ((_Float16*)Atile)[i] = (_Float16)0.f;
  __syncthreads();

  const int w = tid >> 5, lane = tid & 31, ln = lane & 15, hs = lane >> 4;
  const long long* rowIdx = ei;
  const long long* colIdx = ei + E;

  v16h bE1[4][3];
#pragma unroll
  for (int nt = 0; nt < 4; nt++)
#pragma unroll
    for (int s = 0; s < 3; s++)
      bE1[nt][s] = lds_frag_b(&W1t[nt * 16 + ln][0], s * 32, hs);

  for (int t = blockIdx.x * NWAVES + w; t < ntiles; t += gridDim.x * NWAVES) {
    const long long tb = (long long)t * 16;
    {
      long long e = tb + ln; if (e >= E) e = E - 1;
      if (hs == 0) {                                   // xc[row] first
        long long ni = rowIdx[e];
        const unsigned int* src = (const unsigned int*)(F2 + ni * FPITCH);
        unsigned int* dst = (unsigned int*)&Atile[w][ln][0];
#pragma unroll
        for (int i = 0; i < 17; i++) dst[i] = src[i];
        Atile[w][ln][34] = F2[ni * FPITCH + 34];
      } else {                                         // xc[col]
        long long nj = colIdx[e];
        const unsigned int* src = (const unsigned int*)(F2 + nj * FPITCH);
        unsigned int* dst = (unsigned int*)&Atile[w][ln][36];
#pragma unroll
        for (int i = 0; i < 17; i++) dst[i] = src[i];
        Atile[w][ln][70] = F2[nj * FPITCH + 34];
        Atile[w][ln][71] = (_Float16)eattr[e];
      }
    }
    asm volatile("s_wait_dscnt 0" ::: "memory");

    const _Float16* arow = &Atile[w][ln][0];
    v16h a0 = lds_frag_a(arow, 0, hs);
    v16h a1 = lds_frag_a(arow, 32, hs);
    v16h a2 = lds_frag_a(arow, 64, hs);

#pragma unroll
    for (int nt = 0; nt < 4; nt++) {
      v8f c = {0.f, 0.f, 0.f, 0.f, 0.f, 0.f, 0.f, 0.f};
      c = wmma16x16x32(a0, bE1[nt][0], c);
      c = wmma16x16x32(a1, bE1[nt][1], c);
      c = wmma16x16x32(a2, bE1[nt][2], c);
      const int cix = nt * 16 + ln;
      const float bb = B1s[cix];
#pragma unroll
      for (int r = 0; r < 8; r++) {
        float v = c[r] + bb;
        Htile[w][r + 8 * hs][cix] = (_Float16)(v > 0.f ? v : 0.f);
      }
    }
    asm volatile("s_wait_dscnt 0" ::: "memory");

    if (hs == 0) {                                     // N=1 head: per-edge dot
      long long e = tb + ln;
      if (e < E) {
        float s = Be2;
        const _Float16* hp = &Htile[w][ln][0];
#pragma unroll
        for (int k = 0; k < 64; k++) s += (float)hp[k] * We2[k];
        out[e] = 1.f / (1.f + expf(-s));
      }
    }
  }
}

// ---------------------------------------------------------------------------
extern "C" void kernel_launch(void* const* d_in, const int* in_sizes, int n_in,
                              void* d_out, int out_size, void* d_ws, size_t ws_size,
                              hipStream_t stream) {
  (void)n_in; (void)out_size; (void)ws_size;
  const float*     x     = (const float*)d_in[0];
  const long long* ei    = (const long long*)d_in[1];   // int64 (2, E)
  const float*     ea    = (const float*)d_in[2];
  const float*     bn_g  = (const float*)d_in[3];
  const float*     bn_b  = (const float*)d_in[4];
  const float*     w_in1 = (const float*)d_in[5];
  const float*     b_in1 = (const float*)d_in[6];
  const float*     w_in2 = (const float*)d_in[7];
  const float*     b_in2 = (const float*)d_in[8];
  const float*     w_c1  = (const float*)d_in[9];
  const float*     b_c1  = (const float*)d_in[10];
  const float*     w_c2  = (const float*)d_in[11];
  const float*     b_c2  = (const float*)d_in[12];
  const float*     w_e1  = (const float*)d_in[13];
  const float*     b_e1  = (const float*)d_in[14];
  const float*     w_e2  = (const float*)d_in[15];
  const float*     b_e2  = (const float*)d_in[16];
  float*           out   = (float*)d_out;

  const int       N = in_sizes[0] / 3;
  const long long E = (long long)in_sizes[2];   // edge_attr is (E, 1)

  // Workspace layout: [stats 256B][F1 N*80B f16][F2 N*80B f16][Hsum N*32 f32]
  char*      ws    = (char*)d_ws;
  float*     stats = (float*)ws;
  _Float16*  F1    = (_Float16*)(ws + 256);
  _Float16*  F2    = F1 + (size_t)N * FPITCH;
  float*     Hsum  = (float*)(ws + 256 + (size_t)2 * N * FPITCH * sizeof(_Float16));

  const long long totz = (long long)N * HIDN;
  k_zero<<<(int)((totz + 255) / 256), 256, 0, stream>>>(stats, Hsum, totz);

  int sblocks = (N + 255) / 256; if (sblocks > 1024) sblocks = 1024;
  k_stats<<<sblocks, 256, 0, stream>>>(x, stats, N);
  k_finalize<<<1, 32, 0, stream>>>(stats, N);

  k_nodes<<<(N + 255) / 256, 256, 0, stream>>>(x, stats, bn_g, bn_b,
                                               w_in1, b_in1, w_in2, b_in2, F1, N);

  const long long ntiles = (E + 15) / 16;
  const int eblocks = (int)((ntiles + NWAVES - 1) / NWAVES);
  k_edgeconv<<<eblocks, 256, 0, stream>>>(F1, ei, ea, w_c1, b_c1, w_c2, b_c2,
                                          Hsum, E, (int)ntiles);

  k_f2<<<(N + 255) / 256, 256, 0, stream>>>(Hsum, x, stats, bn_g, bn_b, F2, N);

  k_edgeout<<<eblocks, 256, 0, stream>>>(F2, ei, ea, w_e1, b_e1, w_e2, b_e2,
                                         out, E, (int)ntiles);
}
